// EntropyDispatchedLinear_90563680403974
// MI455X (gfx1250) — matrix operand order repaired and checked by reading, not compiled, not run on hardware
//
#include <hip/hip_runtime.h>
#include <stdint.h>

#define TM 32
#define TK 8192
#define TN 28672

typedef _Float16 v16h __attribute__((ext_vector_type(16)));
typedef _Float16 v8h  __attribute__((ext_vector_type(8)));
typedef _Float16 h2   __attribute__((ext_vector_type(2)));
typedef float    v8f  __attribute__((ext_vector_type(8)));

// int8 -> f16 exact dequant via exponent-bias trick:
//   f16_bits(0x6400 | (byte ^ 0x80)) = 1024 + (b + 128); subtract 1152 -> b
// 2 weights in 3 VALU ops (lshl_or, xor, pk_add_f16).
__device__ __forceinline__ h2 dequant2(unsigned b0, unsigned b1) {
  unsigned p = (b0 | (b1 << 16)) ^ 0x64806480u;
  h2 v;
  __builtin_memcpy(&v, &p, sizeof(v));
  const h2 bias = { (_Float16)1152.0f, (_Float16)1152.0f };
  return v - bias;
}

__global__ __launch_bounds__(256)
void EntropyDispatchedLinear_90563680403974_kernel(
    const _Float16* __restrict__ x,
    const signed char* __restrict__ w,
    const float* __restrict__ wscale,
    _Float16* __restrict__ out)
{
  const int lane = threadIdx.x & 31;
  const int wave = threadIdx.x >> 5;
  const int nlo  = lane & 15;    // column / row-within-tile index
  const int hi   = lane >> 4;    // lane half selects K sub-ranges

  const int nb = blockIdx.x * 256 + wave * 32;   // wave's first output column

  // ---- A addressing (x: f16 row-major M x K) in this code's contiguous k order (k = 16*hi + s):
  // lanes 0-15 hold K 0..15, lanes 16-31 hold K 16..31
  const _Float16* xa0 = x + (size_t)nlo * TK + hi * 16;          // M rows 0..15
  const _Float16* xa1 = x + (size_t)(16 + nlo) * TK + hi * 16;   // M rows 16..31

  // ---- B addressing (w: int8 row-major K x N). Same contiguous k order as A:
  // lanes 0-15 hold K 0..15 of the chunk for column nlo, lanes 16-31 hold K 16..31.
  const unsigned char* wb  = (const unsigned char*)w;
  const unsigned char* wp0 = wb + (size_t)(hi * 16) * TN + (size_t)(nb + nlo);
  const unsigned char* wp1 = wp0 + 16;

  v8f acc00 = {}, acc01 = {}, acc10 = {}, acc11 = {};

  #pragma unroll 2
  for (int kb = 0; kb < TK; kb += 32) {
    // A fragments: two 16B contiguous runs each -> global_load_b128 x4
    v8h a0lo = *(const v8h*)(xa0 + kb);
    v8h a0hi = *(const v8h*)(xa0 + kb + 8);
    v8h a1lo = *(const v8h*)(xa1 + kb);
    v8h a1hi = *(const v8h*)(xa1 + kb + 8);
    v16h a0 = __builtin_shufflevector(a0lo, a0hi,
        0,1,2,3,4,5,6,7,8,9,10,11,12,13,14,15);
    v16h a1 = __builtin_shufflevector(a1lo, a1hi,
        0,1,2,3,4,5,6,7,8,9,10,11,12,13,14,15);

    // B fragments: per-lane K-run byte gathers (block-level fully coalesced),
    // register dequant int8 -> f16.
    v16h b0f, b1f;
    {
      const unsigned char* p0 = wp0 + (size_t)kb * TN;
      const unsigned char* p1 = wp1 + (size_t)kb * TN;
      #pragma unroll
      for (int j = 0; j < 8; ++j) {
        unsigned c0 = p0[0];
        unsigned c1 = p0[(size_t)TN];
        unsigned d0 = p1[0];
        unsigned d1 = p1[(size_t)TN];
        p0 += (size_t)(2 * TN);
        p1 += (size_t)(2 * TN);
        h2 e = dequant2(c0, c1);
        h2 f = dequant2(d0, d1);
        b0f[2*j]   = e[0]; b0f[2*j+1] = e[1];
        b1f[2*j]   = f[0]; b1f[2*j+1] = f[1];
      }
    }

    // D = A x B + C  (f32 accumulation)
    acc00 = __builtin_amdgcn_wmma_f32_16x16x32_f16(false, a0, false, b0f,
                                                   (short)0, acc00, false, false);
    acc10 = __builtin_amdgcn_wmma_f32_16x16x32_f16(false, a1, false, b0f,
                                                   (short)0, acc10, false, false);
    acc01 = __builtin_amdgcn_wmma_f32_16x16x32_f16(false, a0, false, b1f,
                                                   (short)0, acc01, false, false);
    acc11 = __builtin_amdgcn_wmma_f32_16x16x32_f16(false, a1, false, b1f,
                                                   (short)0, acc11, false, false);
  }

  // ---- Epilogue: fold per-column dequant scale, convert to f16, store.
  // C/D layout: VGPR r -> (M = r + 8*hi, N = nlo) within the tile.
  const float s0 = wscale[nb + nlo];
  const float s1 = wscale[nb + 16 + nlo];

  _Float16* o00 = out + (size_t)(hi * 8) * TN + (size_t)(nb + nlo);
  _Float16* o01 = o00 + 16;
  _Float16* o10 = out + (size_t)(16 + hi * 8) * TN + (size_t)(nb + nlo);
  _Float16* o11 = o10 + 16;

  #pragma unroll
  for (int r = 0; r < 8; ++r) {
    o00[(size_t)r * TN] = (_Float16)(acc00[r] * s0);
    o01[(size_t)r * TN] = (_Float16)(acc01[r] * s1);
    o10[(size_t)r * TN] = (_Float16)(acc10[r] * s0);
    o11[(size_t)r * TN] = (_Float16)(acc11[r] * s1);
  }
}

extern "C" void kernel_launch(void* const* d_in, const int* in_sizes, int n_in,
                              void* d_out, int out_size, void* d_ws, size_t ws_size,
                              hipStream_t stream) {
  const _Float16*    xptr = (const _Float16*)d_in[0];
  const signed char* wptr = (const signed char*)d_in[1];
  const float*       sptr = (const float*)d_in[2];
  _Float16*          optr = (_Float16*)d_out;

  dim3 grid(TN / 256);   // 112 blocks, 8 waves each: wave owns 32 columns x M=32
  dim3 block(256);
  EntropyDispatchedLinear_90563680403974_kernel<<<grid, block, 0, stream>>>(
      xptr, wptr, sptr, optr);
}
